// NLBlock_51153060496126
// MI455X (gfx1250) — matrix operand order repaired and checked by reading, not compiled, not run on hardware
//
#include <hip/hip_runtime.h>
#include <hip/hip_bf16.h>

// CDNA5 / gfx1250, wave32. WMMA + TDM types per CDNA5_HIP.md / cdna5_isa.
typedef __attribute__((ext_vector_type(16))) __bf16       v16bf;
typedef __attribute__((ext_vector_type(8)))  float        v8f;
typedef __attribute__((ext_vector_type(4)))  unsigned int u32x4;
typedef __attribute__((ext_vector_type(4)))  int          i32x4;
typedef __attribute__((ext_vector_type(8)))  int          i32x8;

#define C_DIM 256
#define CI    128
#define N_DIM 3136
#define N_PAD 3200    // 25 * 128, keeps padded fragment loads in-bounds
#define XT_PITCH 264  // 256 + 8 pad: 528B rows, 16B aligned, conflict-free

// ---------------------------------------------------------------------------
// Tensor Data Mover: 2D tile load Global->LDS (D# per CDNA5 ISA).
// ---------------------------------------------------------------------------
__device__ __forceinline__ void tdm_load_2d(unsigned lds_addr,
                                            unsigned long long gaddr,
                                            i32x8 g1)
{
    u32x4 g0;
    g0[0] = 1u;                                   // count=1, user descriptor
    g0[1] = lds_addr;                             // LDS byte address
    g0[2] = (unsigned)(gaddr & 0xFFFFFFFFu);      // global_addr[31:0]
    g0[3] = (unsigned)((gaddr >> 32) & 0x01FFFFFFu) | (2u << 30); // [56:32]+type=2
    i32x4 gz = {0, 0, 0, 0};                      // groups 2/3 unused (2D)
#if __clang_major__ >= 23
    i32x8 gz8 = {0, 0, 0, 0, 0, 0, 0, 0};
    __builtin_amdgcn_tensor_load_to_lds(g0, g1, gz, gz, gz8, 0);
#else
    __builtin_amdgcn_tensor_load_to_lds(g0, g1, gz, gz, 0);
#endif
}

// ---------------------------------------------------------------------------
// Kernel 0: build the fused weight blocks.
//  - gw_bf: 144x256 bf16 A-matrix = [g_w rows 0..127 | ta row | pb row | 0...]
//    where ta = wf_w[:Ci] @ theta_w, pb = wf_w[Ci:] @ phi_w  (a/bsc are linear
//    in x, so theta/phi collapse entirely).
//  - wz_bf: 256x128 bf16, BN scale folded in. beta2: folded BN shift.
//  - cns: s0 (with wf_b) / t0 constants.
// One block of 256 threads.
// ---------------------------------------------------------------------------
__global__ void nlb_prep(const float* __restrict__ wf_w, const float* __restrict__ wf_b,
                         const float* __restrict__ th_w, const float* __restrict__ th_b,
                         const float* __restrict__ ph_w, const float* __restrict__ ph_b,
                         const float* __restrict__ wz_w, const float* __restrict__ wz_b,
                         const float* __restrict__ gamma, const float* __restrict__ beta,
                         const float* __restrict__ mean,  const float* __restrict__ var,
                         const float* __restrict__ g_w,
                         float* __restrict__ cns, float* __restrict__ beta2,
                         __bf16* __restrict__ wz_bf, __bf16* __restrict__ gw_bf)
{
    const int c = threadIdx.x;                    // 0..255
    float sa = 0.f, sb = 0.f;
    for (int i = 0; i < CI; ++i) {
        sa += wf_w[i]      * th_w[i * C_DIM + c];
        sb += wf_w[CI + i] * ph_w[i * C_DIM + c];
    }
    gw_bf[128 * C_DIM + c] = (__bf16)sa;          // ta row
    gw_bf[129 * C_DIM + c] = (__bf16)sb;          // pb row
    for (int r = 130; r < 144; ++r)               // zero padding rows
        gw_bf[r * C_DIM + c] = (__bf16)0.f;
    if (c < CI)                                   // g_w row c -> bf16
        for (int k = 0; k < C_DIM; ++k)
            gw_bf[c * C_DIM + k] = (__bf16)g_w[c * C_DIM + k];

    const float inv = gamma[c] * rsqrtf(var[c] + 1e-5f);
    beta2[c] = (wz_b[c] - mean[c]) * inv + beta[c];
    for (int k = 0; k < CI; ++k)
        wz_bf[c * CI + k] = (__bf16)(wz_w[c * CI + k] * inv);

    if (c == 0) {
        float s0 = wf_b[0], t0 = 0.f;
        for (int i = 0; i < CI; ++i) {
            s0 += wf_w[i]      * th_b[i];
            t0 += wf_w[CI + i] * ph_b[i];
        }
        cns[0] = s0;
        cns[1] = t0;
    }
}

// ---------------------------------------------------------------------------
// Kernel 1: projection as WMMA GEMM: [g | ta | pb] (144x256) @ x (256 x 128n).
// Phase 1: x tile f32 -> bf16, transposed into LDS (xT[n][c]).
// Phase 2: per wave one 16-wide n-fragment, 9 o-fragments x 8 K-steps.
// o-frags 0..7 -> gT[b][o][n] (+g_b); o-frag 8 rows 0/1 -> s/t (+s0/t0).
// grid (25, 8), 256 threads.
// ---------------------------------------------------------------------------
__global__ void nlb_proj(const float* __restrict__ x, const __bf16* __restrict__ gw_bf,
                         const float* __restrict__ g_b, const float* __restrict__ cns,
                         __bf16* __restrict__ gT, float* __restrict__ sv,
                         float* __restrict__ tv)
{
    __shared__ __bf16 xT[128 * XT_PITCH];         // 66 KB
    const int b = blockIdx.y, n0 = blockIdx.x * 128, tid = threadIdx.x;
    const int lane = tid & 31, wv = tid >> 5;
    const int lm = lane & 15;
    const int half8  = (lane >> 4) * 8;
    const int half16 = (lane >> 4) * 16;

    {   // Phase 1: thread = channel c; zero-fill past N_DIM (last tile only).
        const int c = tid;
        const float4* xr = (const float4*)(x + ((size_t)(b * C_DIM + c)) * N_DIM + n0);
#pragma unroll 8
        for (int q = 0; q < 32; ++q) {
            float4 v;
            if (n0 + 4 * q < N_DIM) v = xr[q];
            else { v.x = 0.f; v.y = 0.f; v.z = 0.f; v.w = 0.f; }
            __bf16* dst = xT + (size_t)(4 * q) * XT_PITCH + c;
            dst[0]            = (__bf16)v.x;
            dst[XT_PITCH]     = (__bf16)v.y;
            dst[2 * XT_PITCH] = (__bf16)v.z;
            dst[3 * XT_PITCH] = (__bf16)v.w;
        }
    }
    __syncthreads();

    const v8f vzero = {0.f, 0.f, 0.f, 0.f, 0.f, 0.f, 0.f, 0.f};
    v8f acc[9];
#pragma unroll
    for (int of = 0; of < 9; ++of) acc[of] = vzero;

    const __bf16* xrow = xT + (size_t)(wv * 16 + lm) * XT_PITCH;
#pragma unroll
    for (int k0 = 0; k0 < C_DIM; k0 += 32) {
        union { v16bf v; uint4 q[2]; } bu;        // B: split-8 K pattern
        bu.q[0] = *(const uint4*)(xrow + k0 + half8);
        bu.q[1] = *(const uint4*)(xrow + k0 + half8 + 16);
#pragma unroll
        for (int of = 0; of < 9; ++of) {          // A: split-8 K pattern
            const __bf16* wr = gw_bf + (size_t)(of * 16 + lm) * C_DIM + k0 + half8;
            union { v16bf v; uint4 q[2]; } au;
            au.q[0] = *(const uint4*)wr;
            au.q[1] = *(const uint4*)(wr + 16);
            acc[of] = __builtin_amdgcn_wmma_f32_16x16x32_bf16(
                false, au.v, false, bu.v, (short)0, acc[of], false, false);
        }
    }

    const int n = n0 + wv * 16 + lm;
    if (n < N_DIM) {
#pragma unroll
        for (int of = 0; of < 8; ++of)
#pragma unroll
            for (int vr = 0; vr < 8; ++vr) {
                const int o = of * 16 + vr + half8;
                gT[((size_t)(b * CI + o)) * N_PAD + n] = (__bf16)(acc[of][vr] + g_b[o]);
            }
        if (lane < 16) {                          // D rows 0/1 of the ta/pb frag
            sv[b * N_PAD + n] = acc[8][0] + cns[0];
            tv[b * N_PAD + n] = acc[8][1] + cns[1];
        }
    }
}

// ---------------------------------------------------------------------------
// Kernel 2: y = (relu(s_i + t_j) / N) @ g  via v_wmma_f32_16x16x32_bf16.
// f is virtual (A-fragments from s/t in registers). g tiles (128k x 64j)
// arrive via double-buffered TDM loads, rows padded 128B->144B (pad interval
// 32 DWORDs, amount 4 DWORDs) for conflict-free ds_load_b128.
// grid (25, 8), 256 threads (8 waves; wave w owns i-rows w*16..+15).
// ---------------------------------------------------------------------------
__global__ void nlb_attn(const float* __restrict__ sv, const float* __restrict__ tv,
                         const __bf16* __restrict__ gT, __bf16* __restrict__ y_nk)
{
    __shared__ __bf16 gs[2][CI * 72];             // two 18 KB buffers, 72-elem pitch
    const int b = blockIdx.y, itile = blockIdx.x, tid = threadIdx.x;
    const int lane = tid & 31, wv = tid >> 5;
    const int lm = lane & 15;
    const int half8  = (lane >> 4) * 8;
    const int half16 = (lane >> 4) * 16;

    const float s_val = sv[b * N_PAD + itile * 128 + wv * 16 + lm];
    const float* tb = tv + b * N_PAD;
    const unsigned long long gbase =
        (unsigned long long)(size_t)(gT + (size_t)b * CI * N_PAD);

    // D# group1: data_size=2B, pad_enable, pad_interval=32 DW, pad_amount=4 DW,
    // tensor 3136 x 128 (stride N_PAD), tile 64 x 128.
    i32x8 g1;
    g1[0] = (1 << 16) | (1 << 20) | (4 << 22) | (3 << 25);
    g1[1] = (int)((N_DIM & 0xFFFF) << 16);        // tensor_dim0 lo16
    g1[2] = (int)((N_DIM >> 16) | ((CI & 0xFFFF) << 16));
    g1[3] = (int)((CI >> 16) | (64 << 16));       // tile_dim0 = 64
    g1[4] = CI;                                   // tile_dim1 = 128
    g1[5] = N_PAD;                                // tensor_dim0_stride
    g1[6] = 0;
    g1[7] = 0;

    const unsigned lds_a[2] = { (unsigned)(size_t)&gs[0][0],
                                (unsigned)(size_t)&gs[1][0] };

    const v8f vzero = {0.f, 0.f, 0.f, 0.f, 0.f, 0.f, 0.f, 0.f};
    v8f acc[8];
#pragma unroll
    for (int kk = 0; kk < 8; ++kk) acc[kk] = vzero;

    if (wv == 0)
        tdm_load_2d(lds_a[0], gbase, g1);         // prologue tile j0=0

    int cur = 0;
    for (int step = 0; step < N_DIM / 64; ++step) {
        const int j0 = step * 64;
        if (wv == 0)
            __builtin_amdgcn_s_wait_tensorcnt(0); // buf[cur] landed
        __syncthreads();                          // visible to all waves;
                                                  // prev readers of buf[cur^1] done
        if (wv == 0 && step + 1 < N_DIM / 64)
            tdm_load_2d(lds_a[cur ^ 1], gbase + 2ull * (j0 + 64), g1);

        const __bf16* gsc = &gs[cur][0];
#pragma unroll
        for (int jc = 0; jc < 2; ++jc) {          // two 32-wide j chunks
            const int jb = j0 + jc * 32;
            // A fragment of f: relu(s_i + t_j); K map: e<8 -> half8+e,
            // e>=8 -> 16+half8+(e-8)
            const float4 t0 = *(const float4*)(tb + jb + half8);
            const float4 t1 = *(const float4*)(tb + jb + half8 + 4);
            const float4 t2 = *(const float4*)(tb + jb + 16 + half8);
            const float4 t3 = *(const float4*)(tb + jb + 20 + half8);
            v16bf a;
            a[0]  = (__bf16)fmaxf(s_val + t0.x, 0.f);
            a[1]  = (__bf16)fmaxf(s_val + t0.y, 0.f);
            a[2]  = (__bf16)fmaxf(s_val + t0.z, 0.f);
            a[3]  = (__bf16)fmaxf(s_val + t0.w, 0.f);
            a[4]  = (__bf16)fmaxf(s_val + t1.x, 0.f);
            a[5]  = (__bf16)fmaxf(s_val + t1.y, 0.f);
            a[6]  = (__bf16)fmaxf(s_val + t1.z, 0.f);
            a[7]  = (__bf16)fmaxf(s_val + t1.w, 0.f);
            a[8]  = (__bf16)fmaxf(s_val + t2.x, 0.f);
            a[9]  = (__bf16)fmaxf(s_val + t2.y, 0.f);
            a[10] = (__bf16)fmaxf(s_val + t2.z, 0.f);
            a[11] = (__bf16)fmaxf(s_val + t2.w, 0.f);
            a[12] = (__bf16)fmaxf(s_val + t3.x, 0.f);
            a[13] = (__bf16)fmaxf(s_val + t3.y, 0.f);
            a[14] = (__bf16)fmaxf(s_val + t3.z, 0.f);
            a[15] = (__bf16)fmaxf(s_val + t3.w, 0.f);

#pragma unroll
            for (int kk = 0; kk < 8; ++kk) {
                const int kcol = kk * 16 + lm;
                union { v16bf v; uint4 q[2]; } bu;
                bu.q[0] = *(const uint4*)(gsc + kcol * 72 + jc * 32 + half8);
                bu.q[1] = *(const uint4*)(gsc + kcol * 72 + jc * 32 + half8 + 16);
                acc[kk] = __builtin_amdgcn_wmma_f32_16x16x32_bf16(
                    false, a, false, bu.v, (short)0, acc[kk], false, false);
            }
        }
        cur ^= 1;
    }

    // Store y as (N_PAD x CI) bf16 row-major (= K3's B-fragment layout), with
    // the 1/N scale folded in. Branch once for the single partial i-tile.
    const float inv_n = 1.0f / (float)N_DIM;
    const int ibase = itile * 128 + wv * 16 + half8;
    __bf16* ydst = y_nk + ((size_t)b * N_PAD + ibase) * CI + lm;
    if (ibase + 8 <= N_DIM) {
#pragma unroll
        for (int kk = 0; kk < 8; ++kk)
#pragma unroll
            for (int vr = 0; vr < 8; ++vr)
                ydst[(size_t)vr * CI + kk * 16] = (__bf16)(acc[kk][vr] * inv_n);
    } else {
#pragma unroll
        for (int kk = 0; kk < 8; ++kk)
#pragma unroll
            for (int vr = 0; vr < 8; ++vr)
                if (ibase + vr < N_DIM)
                    ydst[(size_t)vr * CI + kk * 16] = (__bf16)(acc[kk][vr] * inv_n);
    }
}

// ---------------------------------------------------------------------------
// Kernel 3: z = (BN-folded wz) @ y^T + beta2 + x, via WMMA.
// Per block: 256 o-rows x 64 n-cols. y tile (64 x 128 bf16) staged once in
// LDS via TDM (row pitch 128 -> 136 elems: interval 64 DW, amount 4 DW).
// grid (49, 8), 256 threads.
// ---------------------------------------------------------------------------
__global__ void nlb_out(const __bf16* __restrict__ wz_bf, const __bf16* __restrict__ y_nk,
                        const float* __restrict__ beta2, const float* __restrict__ x,
                        float* __restrict__ out)
{
    __shared__ __bf16 ys[64 * 136];               // 17 KB, 136-elem (272B) pitch
    const int b = blockIdx.y, nt = blockIdx.x, tid = threadIdx.x;
    const int lane = tid & 31, wv = tid >> 5;
    const int lm = lane & 15;
    const int half8  = (lane >> 4) * 8;
    const int half16 = (lane >> 4) * 16;
    const int o0 = wv * 32, nbase = nt * 64;

    if (wv == 0) {
        // D# group1: data_size=2B, pad_enable, pad_interval=64 DW, pad=4 DW,
        // tensor 128 x N_PAD (stride CI), tile 128 x 64.
        i32x8 g1;
        g1[0] = (1 << 16) | (1 << 20) | (5 << 22) | (3 << 25);
        g1[1] = (int)((CI & 0xFFFF) << 16);       // tensor_dim0 lo16
        g1[2] = (int)((CI >> 16) | ((N_PAD & 0xFFFF) << 16));
        g1[3] = (int)((N_PAD >> 16) | (CI << 16)); // tile_dim0 = 128
        g1[4] = 64;                               // tile_dim1 = 64
        g1[5] = CI;                               // tensor_dim0_stride
        g1[6] = 0;
        g1[7] = 0;
        const unsigned long long ga =
            (unsigned long long)(size_t)(y_nk + ((size_t)b * N_PAD + nbase) * CI);
        tdm_load_2d((unsigned)(size_t)&ys[0], ga, g1);
        __builtin_amdgcn_s_wait_tensorcnt(0);
    }
    __syncthreads();

    const v8f vzero = {0.f, 0.f, 0.f, 0.f, 0.f, 0.f, 0.f, 0.f};
    v8f acc[2][4];
#pragma unroll
    for (int oi = 0; oi < 2; ++oi)
#pragma unroll
        for (int ni = 0; ni < 4; ++ni) acc[oi][ni] = vzero;

#pragma unroll
    for (int k0 = 0; k0 < CI; k0 += 32) {
        union { v16bf v; uint4 q[2]; } au[2];
#pragma unroll
        for (int oi = 0; oi < 2; ++oi) {          // A: split-8 K pattern
            const __bf16* wr = wz_bf + (size_t)(o0 + oi * 16 + lm) * CI + k0 + half8;
            au[oi].q[0] = *(const uint4*)wr;
            au[oi].q[1] = *(const uint4*)(wr + 16);
        }
        union { v16bf v; uint4 q[2]; } bu[4];
#pragma unroll
        for (int ni = 0; ni < 4; ++ni) {          // B: split-8 K pattern
            const __bf16* yr = ys + (size_t)(ni * 16 + lm) * 136 + k0 + half8;
            bu[ni].q[0] = *(const uint4*)yr;
            bu[ni].q[1] = *(const uint4*)(yr + 16);
        }
#pragma unroll
        for (int ni = 0; ni < 4; ++ni) {
            acc[0][ni] = __builtin_amdgcn_wmma_f32_16x16x32_bf16(
                false, au[0].v, false, bu[ni].v, (short)0, acc[0][ni], false, false);
            acc[1][ni] = __builtin_amdgcn_wmma_f32_16x16x32_bf16(
                false, au[1].v, false, bu[ni].v, (short)0, acc[1][ni], false, false);
        }
    }

#pragma unroll
    for (int oi = 0; oi < 2; ++oi)
#pragma unroll
        for (int ni = 0; ni < 4; ++ni)
#pragma unroll
            for (int vr = 0; vr < 8; ++vr) {
                const int o = o0 + oi * 16 + vr + half8;
                const int n = nbase + ni * 16 + lm;
                const size_t idx = ((size_t)(b * C_DIM + o)) * N_DIM + n;
                out[idx] = acc[oi][ni][vr] + beta2[o] + x[idx];
            }
}

// ---------------------------------------------------------------------------
extern "C" void kernel_launch(void* const* d_in, const int* in_sizes, int n_in,
                              void* d_out, int out_size, void* d_ws, size_t ws_size,
                              hipStream_t stream)
{
    const float* x     = (const float*)d_in[0];
    const float* g_w   = (const float*)d_in[1];
    const float* g_b   = (const float*)d_in[2];
    const float* th_w  = (const float*)d_in[3];
    const float* th_b  = (const float*)d_in[4];
    const float* ph_w  = (const float*)d_in[5];
    const float* ph_b  = (const float*)d_in[6];
    const float* wf_w  = (const float*)d_in[7];
    const float* wf_b  = (const float*)d_in[8];
    const float* wz_w  = (const float*)d_in[9];
    const float* wz_b  = (const float*)d_in[10];
    const float* gamma = (const float*)d_in[11];
    const float* beta  = (const float*)d_in[12];
    const float* mean  = (const float*)d_in[13];
    const float* var   = (const float*)d_in[14];
    float* out = (float*)d_out;

    // Workspace layout (all offsets 256B-aligned); total ~13.5 MB.
    char* ws = (char*)d_ws;
    float*  cns   = (float*)(ws + 0);         // 2 f32 (pad to 256B)
    float*  beta2 = (float*)(ws + 256);       // 256 f32
    __bf16* wz_bf = (__bf16*)(ws + 2048);     // 256*128 bf16 = 64 KB
    __bf16* gw_bf = (__bf16*)(ws + 67584);    // 144*256 bf16 = 72 KB
    float*  sv    = (float*)(ws + 141312);    // 8*3200 f32
    float*  tv    = (float*)(ws + 243712);    // 8*3200 f32
    __bf16* gT    = (__bf16*)(ws + 346112);   // 8*128*3200 bf16 = 6.25 MB
    __bf16* y_nk  = (__bf16*)(ws + 6899712);  // 8*3200*128 bf16 = 6.25 MB

    nlb_prep<<<dim3(1), dim3(256), 0, stream>>>(
        wf_w, wf_b, th_w, th_b, ph_w, ph_b, wz_w, wz_b,
        gamma, beta, mean, var, g_w, cns, beta2, wz_bf, gw_bf);

    nlb_proj<<<dim3(25, 8), dim3(256), 0, stream>>>(
        x, gw_bf, g_b, cns, gT, sv, tv);

    nlb_attn<<<dim3(25, 8), dim3(256), 0, stream>>>(sv, tv, gT, y_nk);

    nlb_out<<<dim3(49, 8), dim3(256), 0, stream>>>(wz_bf, y_nk, beta2, x, out);
}
